// CELPNet_47012712022159
// MI455X (gfx1250) — matrix operand order repaired and checked by reading, not compiled, not run on hardware
//
#include <hip/hip_runtime.h>
#include <hip/hip_bf16.h>

// ---------------------------------------------------------------------------
// Model constants
// ---------------------------------------------------------------------------
#define BATCH     64
#define TT        104      // NB_FRAMES + 4
#define NBF       100
#define STEPS     400      // NBF * NSUB
#define FEAT      20
#define PEMB      64
#define COND      256
#define SUBN      40
#define FRAME_N   160

typedef _Float16 h16 __attribute__((ext_vector_type(16)));
typedef _Float16 h8  __attribute__((ext_vector_type(8)));
typedef float    f8  __attribute__((ext_vector_type(8)));

__device__ __forceinline__ f8 zero8() {
    f8 z = {0.f, 0.f, 0.f, 0.f, 0.f, 0.f, 0.f, 0.f};
    return z;
}

__device__ __forceinline__ f8 wmma16(h16 a, h16 b, f8 c) {
    return __builtin_amdgcn_wmma_f32_16x16x32_f16(false, a, false, b,
                                                  (short)0, c, false, false);
}

// Branch-free tanh/sigmoid: v_exp_f32 + v_rcp_f32, no EXEC divergence.
// (libm tanhf expands to two exec-masked paths that both run with mixed
//  lanes -- deadly on the 400-step latency chain.)
__device__ __forceinline__ float fast_tanh(float x) {
    float a = __builtin_fabsf(x);
    float e = __expf(-2.f * a);
    float t = (1.f - e) * __builtin_amdgcn_rcpf(1.f + e);
    return __builtin_copysignf(t, x);
}

__device__ __forceinline__ float fast_sig(float x) {
    return __builtin_amdgcn_rcpf(1.f + __expf(-x));
}

// A-operand fragment (16x32, f16): lane = row (mod 16), element e maps to
// k = (e<8 ? e : e+8) + 8*(lane>=16).  Source rows are row-major f16 with
// row stride `lda` halves, so each fragment is two 16B contiguous loads.
__device__ __forceinline__ h16 ld_a(const _Float16* base, int lda, int row0,
                                    int k0, int lane) {
    int r  = row0 + (lane & 15);
    int kb = k0 + ((lane >> 4) << 3);
    const _Float16* p = base + (size_t)r * lda + kb;
    h8 lo = *(const h8*)(p);
    h8 hi = *(const h8*)(p + 16);
    h16 o;
#pragma unroll
    for (int e = 0; e < 8; ++e) { o[e] = lo[e]; o[e + 8] = hi[e]; }
    return o;
}

// B-operand fragment from our packed buffer: one v16h per (tile, lane).
__device__ __forceinline__ h16 ld_b(const _Float16* __restrict__ pk, int nt,
                                    int kb, int nkb, int lane) {
    return *(const h16*)(pk + ((size_t)(nt * nkb + kb) * 32 + lane) * 16);
}

// B packing map: lane = column (mod 16), element e maps to
// k = (e<8 ? e : e+8) + 8*(lane>=16)   (same k order as the A fragment)
__device__ __forceinline__ int kmapB(int e, int lane) {
    return (e < 8 ? e : e + 8) + ((lane >> 4) << 3);
}

// ---------------------------------------------------------------------------
// Weight packing kernels (f32 -> packed f16, fragment-native layout)
// ---------------------------------------------------------------------------
__global__ void pack_plain(const float* __restrict__ W, int K, int N, int ld,
                           _Float16* __restrict__ dst, int Kp, int Np) {
    int idx = blockIdx.x * blockDim.x + threadIdx.x;
    int nkb = Kp >> 5, ntN = Np >> 4;
    int total = nkb * ntN * 32;
    if (idx >= total) return;
    int lane = idx & 31, tile = idx >> 5;
    int kb = tile % nkb, nt = tile / nkb;
    int col = nt * 16 + (lane & 15);
    _Float16* p = dst + ((size_t)(nt * nkb + kb) * 32 + lane) * 16;
#pragma unroll
    for (int e = 0; e < 16; ++e) {
        int k = kb * 32 + kmapB(e, lane);
        float v = (k < K && col < N) ? W[(size_t)k * ld + col] : 0.f;
        p[e] = (_Float16)v;
    }
}

// conv weight (O=256, I=256, k=3) -> GEMM weight W[k*256+i][o], K=768, N=256
__global__ void pack_conv(const float* __restrict__ cw,
                          _Float16* __restrict__ dst) {
    int idx = blockIdx.x * blockDim.x + threadIdx.x;
    const int nkb = 24, ntN = 16;
    int total = nkb * ntN * 32;
    if (idx >= total) return;
    int lane = idx & 31, tile = idx >> 5;
    int kb = tile % nkb, nt = tile / nkb;
    int col = nt * 16 + (lane & 15);
    _Float16* p = dst + ((size_t)(nt * nkb + kb) * 32 + lane) * 16;
#pragma unroll
    for (int e = 0; e < 16; ++e) {
        int k = kb * 32 + kmapB(e, lane);
        int kk = k >> 8, i = k & 255;
        p[e] = (_Float16)cw[(size_t)col * 768 + i * 3 + kk];
    }
}

// W_pre (352x256) = rows [sd1_w[0:256] ; sd1_w[296:336] ; sd1_w[336:376] ; 0]
__global__ void pack_pre(const float* __restrict__ sd1w,
                         _Float16* __restrict__ dst) {
    int idx = blockIdx.x * blockDim.x + threadIdx.x;
    const int nkb = 11, ntN = 16;
    int total = nkb * ntN * 32;
    if (idx >= total) return;
    int lane = idx & 31, tile = idx >> 5;
    int kb = tile % nkb, nt = tile / nkb;
    int col = nt * 16 + (lane & 15);
    _Float16* p = dst + ((size_t)(nt * nkb + kb) * 32 + lane) * 16;
#pragma unroll
    for (int e = 0; e < 16; ++e) {
        int k = kb * 32 + kmapB(e, lane);
        int row = -1;
        if (k < 256) row = k;                       // cond part
        else if (k < 296) row = 296 + (k - 256);    // phase_real part
        else if (k < 336) row = 336 + (k - 296);    // phase_imag part
        float v = (row >= 0) ? sd1w[(size_t)row * 256 + col] : 0.f;
        p[e] = (_Float16)v;
    }
}

// ---------------------------------------------------------------------------
// Activation builders
// ---------------------------------------------------------------------------
// X1 (6656 x 96 f16): [features(20) | pembed[period](64) | 0(12)]
__global__ void build_x1(const float* __restrict__ feat,
                         const int* __restrict__ period,
                         const float* __restrict__ pembed,
                         _Float16* __restrict__ X1) {
    int idx = blockIdx.x * blockDim.x + threadIdx.x;
    const int total = BATCH * TT * 96;
    if (idx >= total) return;
    int r = idx / 96, c = idx % 96;
    float v = 0.f;
    if (c < FEAT) v = feat[(size_t)r * FEAT + c];
    else if (c < FEAT + PEMB) {
        int pd = period[r];
        v = pembed[(size_t)pd * PEMB + (c - FEAT)];
    }
    X1[idx] = (_Float16)v;
}

// Phase embedding into Xpre columns [256..335] (+ zero pad [336..351])
__global__ void phase_kernel(const int* __restrict__ period,
                             const float* __restrict__ rand_shift,
                             _Float16* __restrict__ Xpre) {
    int b = blockIdx.x;
    __shared__ float cum[NBF];
    __shared__ float w0s[NBF];
    if (threadIdx.x == 0) {
        const float twopi = 6.28318530717958647692f;
        float acc = 0.f, w0prev = 0.f;
        for (int s = 0; s < NBF; ++s) {
            float p  = (float)period[b * TT + 3 + s];
            float w0 = twopi / p;
            float w0sh = (s == 0) ? twopi * rand_shift[b] / (float)FRAME_N
                                  : w0prev;
            acc += (float)FRAME_N * w0sh;
            cum[s] = acc;
            w0s[s] = w0;
            w0prev = w0;
        }
    }
    __syncthreads();
    for (int i = threadIdx.x; i < STEPS * SUBN; i += blockDim.x) {
        int step = i / SUBN, j = i % SUBN;
        int f = step >> 2, sub = step & 3;
        float ph = cum[f] + w0s[f] * (float)(sub * SUBN + j);
        size_t ro = ((size_t)b * STEPS + step) * 352;
        Xpre[ro + 256 + j] = (_Float16)cosf(ph);
        Xpre[ro + 296 + j] = (_Float16)sinf(ph);
    }
    for (int i = threadIdx.x; i < STEPS * 16; i += blockDim.x) {
        int step = i / 16, c = i % 16;
        Xpre[((size_t)b * STEPS + step) * 352 + 336 + c] = (_Float16)0.f;
    }
}

// Replicate cond rows (x4 sub-frames) into Xpre columns [0..255]
__global__ void fill_cond(const _Float16* __restrict__ cond,
                          _Float16* __restrict__ Xpre) {
    size_t idx = (size_t)blockIdx.x * blockDim.x + threadIdx.x;
    const size_t total = (size_t)BATCH * STEPS * 256;
    if (idx >= total) return;
    int c = (int)(idx & 255);
    int r = (int)(idx >> 8);
    int b = r / STEPS, step = r % STEPS, f = step >> 2;
    Xpre[(size_t)r * 352 + c] = cond[((size_t)b * NBF + f) * 256 + c];
}

// ---------------------------------------------------------------------------
// Generic WMMA GEMM:  out = act(A_f16 @ Bpacked + bias)
// A row r lives at A + (r/rowsPerGroup)*groupStride + (r%rowsPerGroup)*rowStride
// (groups let the conv layers read K=768 contiguous windows in-place).
// PERM=1: f32 output written step-major: row (b*STEPS+s) -> [s][b] (locality
// for the recurrent kernel, which consumes one 64KB step-slice at a time).
// ---------------------------------------------------------------------------
template <int TANH, int OUTF16, int PERM>
__global__ void gemm_wmma(const _Float16* __restrict__ A, int rowsPerGroup,
                          long long groupStride, int rowStride,
                          const _Float16* __restrict__ Bp, int Kp,
                          const float* __restrict__ bias, int M, int N,
                          void* __restrict__ outv, int ldo) {
    int lane = threadIdx.x & 31;
    int wave = (blockIdx.x * blockDim.x + threadIdx.x) >> 5;
    int nWaves = (gridDim.x * blockDim.x) >> 5;
    int Mt = M >> 4, Nt = N >> 4, nkb = Kp >> 5;
    int hi8 = (lane >> 4) << 3;
    for (int t = wave; t < Mt * Nt; t += nWaves) {
        int mt = t % Mt, nt = t / Mt;
        int row = mt * 16 + (lane & 15);
        long long aoff = (long long)(row / rowsPerGroup) * groupStride +
                         (long long)(row % rowsPerGroup) * rowStride;
        const _Float16* Ar = A + aoff;
        f8 acc = zero8();
        for (int kb = 0; kb < nkb; ++kb) {
            int kbase = kb * 32 + hi8;
            h8 lo = *(const h8*)(Ar + kbase);
            h8 hv = *(const h8*)(Ar + kbase + 16);
            h16 a;
#pragma unroll
            for (int e = 0; e < 8; ++e) { a[e] = lo[e]; a[e + 8] = hv[e]; }
            h16 b = ld_b(Bp, nt, kb, nkb, lane);
            acc = wmma16(a, b, acc);
        }
        int col = nt * 16 + (lane & 15);
        float bv = bias[col];
#pragma unroll
        for (int j = 0; j < 8; ++j) {
            int m = mt * 16 + j + hi8;
            float v = acc[j] + bv;
            if (TANH) v = fast_tanh(v);
            if (OUTF16) {
                ((_Float16*)outv)[(size_t)m * ldo + col] = (_Float16)v;
            } else if (PERM) {
                int bb = m / STEPS, ss = m % STEPS;
                ((float*)outv)[((size_t)ss * BATCH + bb) * 256 + col] = v;
            } else {
                ((float*)outv)[(size_t)m * ldo + col] = v;
            }
        }
    }
}

// ---------------------------------------------------------------------------
// Persistent single-WGP recurrent kernel: 400 steps of
//   tt  = tanh(pre[step] + prev @ Wp)
//   tt  = tanh(tt @ sd2 + b2)
//   h1..h3 = GRU chain (all gates via WMMA, fused elementwise combine)
//   out = tanh(h3 @ Wout + bo)   -> d_out + prev feedback
// State lives in 5 x 32KB LDS banks rotated each step (160KB total).
// ---------------------------------------------------------------------------
__global__ __launch_bounds__(512) void recurrent_kernel(
    const _Float16* __restrict__ wpp,   // prev weights  64x256  (nkb=2)
    const _Float16* __restrict__ sd2p,  // 256x256 (nkb=8)
    const _Float16* __restrict__ wi0, const _Float16* __restrict__ wh0,
    const _Float16* __restrict__ wi1, const _Float16* __restrict__ wh1,
    const _Float16* __restrict__ wi2, const _Float16* __restrict__ wh2,
    const _Float16* __restrict__ outp,  // 256x48 (nkb=8)
    const float* __restrict__ pre,      // step-major: [STEPS][BATCH][256] f32
    const float* __restrict__ sd2_b,
    const float* __restrict__ bi0, const float* __restrict__ bh0,
    const float* __restrict__ bi1, const float* __restrict__ bh1,
    const float* __restrict__ bi2, const float* __restrict__ bh2,
    const float* __restrict__ out_b,
    float* __restrict__ out) {
    __shared__ __align__(16) _Float16 smem[5 * 16384];  // 160 KB
    const int tid = threadIdx.x, lane = tid & 31, w = tid >> 5;
    const int NW = 16;  // waves in block
    const int hi8 = (lane >> 4) << 3;

    for (int i = tid; i < 5 * 16384; i += 512) smem[i] = (_Float16)0.f;
    __syncthreads();

    const _Float16* wis[3] = {wi0, wi1, wi2};
    const _Float16* whs[3] = {wh0, wh1, wh2};
    const float* bis[3] = {bi0, bi1, bi2};
    const float* bhs[3] = {bh0, bh1, bh2};

    int hidx0 = 0, hidx1 = 1, hidx2 = 2, pA = 3;
    const int pB = 4;  // fixed: holds tt, tail hosts `prev` between steps

    for (int step = 0; step < STEPS; ++step) {
        _Float16* prevp = smem + (size_t)pB * 16384 + 12288;  // 64x64 f16
        _Float16* t0    = smem + (size_t)pA * 16384;          // sd1 out
        _Float16* ttp   = smem + (size_t)pB * 16384;          // sd2 out
        const float* preS = pre + (size_t)step * (BATCH * 256);

        // prefetch next step's 64KB pre-slice (512 thr x 128B)
        if (step + 1 < STEPS)
            __builtin_prefetch(pre + (size_t)(step + 1) * (BATCH * 256) +
                                   (size_t)tid * 32, 0, 0);

        // ---- stage A: tt0 = tanh(pre + prev @ Wp) ----
        for (int t = w; t < 64; t += NW) {
            int mt = t & 3, nt = t >> 2;
            f8 acc = zero8();
#pragma unroll
            for (int kb = 0; kb < 2; ++kb) {
                h16 a = ld_a(prevp, 64, mt * 16, kb * 32, lane);
                h16 b = ld_b(wpp, nt, kb, 2, lane);
                acc = wmma16(a, b, acc);
            }
            int col = nt * 16 + (lane & 15);
#pragma unroll
            for (int j = 0; j < 8; ++j) {
                int m = mt * 16 + j + hi8;
                float v = acc[j] + preS[(size_t)m * 256 + col];
                t0[m * 256 + col] = (_Float16)fast_tanh(v);
            }
        }
        __syncthreads();

        // ---- stage B: tt = tanh(tt0 @ sd2 + b2) ----
        for (int t = w; t < 64; t += NW) {
            int mt = t & 3, nt = t >> 2;
            f8 acc = zero8();
#pragma unroll
            for (int kb = 0; kb < 8; ++kb) {
                h16 a = ld_a(t0, 256, mt * 16, kb * 32, lane);
                h16 b = ld_b(sd2p, nt, kb, 8, lane);
                acc = wmma16(a, b, acc);
            }
            int col = nt * 16 + (lane & 15);
            float bv = sd2_b[col];
#pragma unroll
            for (int j = 0; j < 8; ++j) {
                int m = mt * 16 + j + hi8;
                ttp[m * 256 + col] = (_Float16)fast_tanh(acc[j] + bv);
            }
        }
        __syncthreads();

        // ---- GRU chain (bank rotation: write h_new into a free bank) ----
        int harr0 = hidx0, harr1 = hidx1, harr2 = hidx2;
        int Xi = pB, dst = pA;
#pragma unroll
        for (int g = 0; g < 3; ++g) {
            int Hg = (g == 0) ? harr0 : (g == 1) ? harr1 : harr2;
            _Float16* X = smem + (size_t)Xi * 16384;
            _Float16* H = smem + (size_t)Hg * 16384;
            _Float16* D = smem + (size_t)dst * 16384;
            const _Float16* wi = wis[g];
            const _Float16* wh = whs[g];
            const float* bi = bis[g];
            const float* bh = bhs[g];
            for (int t = w; t < 64; t += NW) {
                int mt = t & 3, ct = t >> 2;
                f8 ar = zero8(), az = zero8(), an = zero8();
                f8 hr = zero8(), hz = zero8(), hn = zero8();
#pragma unroll
                for (int kb = 0; kb < 8; ++kb) {
                    h16 ax = ld_a(X, 256, mt * 16, kb * 32, lane);
                    h16 ah = ld_a(H, 256, mt * 16, kb * 32, lane);
                    ar = wmma16(ax, ld_b(wi, ct, kb, 8, lane), ar);
                    az = wmma16(ax, ld_b(wi, 16 + ct, kb, 8, lane), az);
                    an = wmma16(ax, ld_b(wi, 32 + ct, kb, 8, lane), an);
                    hr = wmma16(ah, ld_b(wh, ct, kb, 8, lane), hr);
                    hz = wmma16(ah, ld_b(wh, 16 + ct, kb, 8, lane), hz);
                    hn = wmma16(ah, ld_b(wh, 32 + ct, kb, 8, lane), hn);
                }
                int col = ct * 16 + (lane & 15);
                float br = bi[col], bz = bi[256 + col], bn = bi[512 + col];
                float cr = bh[col], cz = bh[256 + col], cn = bh[512 + col];
#pragma unroll
                for (int j = 0; j < 8; ++j) {
                    int m = mt * 16 + j + hi8;
                    float rg = fast_sig(ar[j] + br + hr[j] + cr);
                    float zg = fast_sig(az[j] + bz + hz[j] + cz);
                    float ng = fast_tanh(an[j] + bn + rg * (hn[j] + cn));
                    float ho = (float)H[m * 256 + col];
                    D[m * 256 + col] =
                        (_Float16)((1.f - zg) * ng + zg * ho);
                }
            }
            __syncthreads();
            if (g == 0) harr0 = dst;
            else if (g == 1) harr1 = dst;
            else harr2 = dst;
            Xi = dst;
            dst = Hg;
        }

        // ---- stage D: out = tanh(h3 @ Wout + bo) ----
        _Float16* H3 = smem + (size_t)harr2 * 16384;
        for (int t = w; t < 12; t += NW) {
            int mt = t & 3, nt = t >> 2;
            f8 acc = zero8();
#pragma unroll
            for (int kb = 0; kb < 8; ++kb) {
                h16 a = ld_a(H3, 256, mt * 16, kb * 32, lane);
                h16 b = ld_b(outp, nt, kb, 8, lane);
                acc = wmma16(a, b, acc);
            }
            int col = nt * 16 + (lane & 15);
            if (col < SUBN) {
                float ob = out_b[col];
#pragma unroll
                for (int j = 0; j < 8; ++j) {
                    int m = mt * 16 + j + hi8;
                    float v = fast_tanh(acc[j] + ob);
                    out[(size_t)m * (STEPS * SUBN) + step * SUBN + col] = v;
                    prevp[m * 64 + col] = (_Float16)v;
                    // prev cols >= 40 may hold stale tt data; Wp rows >= 40
                    // are packed as zeros so they never contribute.
                }
            }
        }
        hidx0 = harr0; hidx1 = harr1; hidx2 = harr2; pA = dst;
        __syncthreads();
    }

    // ---- final hidden states: out[1024000 + g*16384 + m*256 + c] ----
    const size_t base = (size_t)BATCH * STEPS * SUBN;
    int hfin[3] = {hidx0, hidx1, hidx2};
#pragma unroll
    for (int g = 0; g < 3; ++g) {
        _Float16* H = smem + (size_t)hfin[g] * 16384;
        for (int i = tid; i < 16384; i += 512)
            out[base + (size_t)g * 16384 + i] = (float)H[i];
    }
}

// ---------------------------------------------------------------------------
// Host launcher
// ---------------------------------------------------------------------------
extern "C" void kernel_launch(void* const* d_in, const int* in_sizes, int n_in,
                              void* d_out, int out_size, void* d_ws,
                              size_t ws_size, hipStream_t stream) {
    (void)in_sizes; (void)n_in; (void)out_size; (void)ws_size;

    const float* features = (const float*)d_in[0];
    const int*   period   = (const int*)d_in[1];
    const float* rshift   = (const float*)d_in[3];
    const float* pembed   = (const float*)d_in[4];
    const float* fd1_w = (const float*)d_in[5];  const float* fd1_b = (const float*)d_in[6];
    const float* c1_w  = (const float*)d_in[7];  const float* c1_b  = (const float*)d_in[8];
    const float* c2_w  = (const float*)d_in[9];  const float* c2_b  = (const float*)d_in[10];
    const float* fd2_w = (const float*)d_in[11]; const float* fd2_b = (const float*)d_in[12];
    const float* sd1_w = (const float*)d_in[13]; const float* sd1_b = (const float*)d_in[14];
    const float* sd2_w = (const float*)d_in[15]; const float* sd2_b = (const float*)d_in[16];
    const float* g1_wi = (const float*)d_in[17]; const float* g1_bi = (const float*)d_in[18];
    const float* g1_wh = (const float*)d_in[19]; const float* g1_bh = (const float*)d_in[20];
    const float* g2_wi = (const float*)d_in[21]; const float* g2_bi = (const float*)d_in[22];
    const float* g2_wh = (const float*)d_in[23]; const float* g2_bh = (const float*)d_in[24];
    const float* g3_wi = (const float*)d_in[25]; const float* g3_bi = (const float*)d_in[26];
    const float* g3_wh = (const float*)d_in[27]; const float* g3_bh = (const float*)d_in[28];
    const float* out_w = (const float*)d_in[29]; const float* out_b = (const float*)d_in[30];
    float* out = (float*)d_out;

    char* ws = (char*)d_ws;
    size_t cur = 0;
    auto alloc = [&](size_t bytes) -> void* {
        cur = (cur + 255) & ~(size_t)255;
        void* p = ws + cur;
        cur += bytes;
        return p;
    };
    auto packP = [&](const float* W, int K, int N, int ld, int Kp,
                     int Np) -> _Float16* {
        _Float16* dst = (_Float16*)alloc((size_t)Kp * Np * 2);
        int total = (Kp / 32) * (Np / 16) * 32;
        pack_plain<<<(total + 255) / 256, 256, 0, stream>>>(W, K, N, ld, dst,
                                                            Kp, Np);
        return dst;
    };

    // ---- pack weights ----
    _Float16* fd1p = packP(fd1_w, 84, 256, 256, 96, 256);
    _Float16* c1p  = (_Float16*)alloc((size_t)768 * 256 * 2);
    pack_conv<<<(24 * 16 * 32 + 255) / 256, 256, 0, stream>>>(c1_w, c1p);
    _Float16* c2p  = (_Float16*)alloc((size_t)768 * 256 * 2);
    pack_conv<<<(24 * 16 * 32 + 255) / 256, 256, 0, stream>>>(c2_w, c2p);
    _Float16* fd2p = packP(fd2_w, 256, 256, 256, 256, 256);
    _Float16* prep = (_Float16*)alloc((size_t)352 * 256 * 2);
    pack_pre<<<(11 * 16 * 32 + 255) / 256, 256, 0, stream>>>(sd1_w, prep);
    _Float16* wpp  = packP(sd1_w + 256 * 256, 40, 256, 256, 64, 256);
    _Float16* sd2p = packP(sd2_w, 256, 256, 256, 256, 256);
    _Float16* wi0p = packP(g1_wi, 256, 768, 768, 256, 768);
    _Float16* wh0p = packP(g1_wh, 256, 768, 768, 256, 768);
    _Float16* wi1p = packP(g2_wi, 256, 768, 768, 256, 768);
    _Float16* wh1p = packP(g2_wh, 256, 768, 768, 256, 768);
    _Float16* wi2p = packP(g3_wi, 256, 768, 768, 256, 768);
    _Float16* wh2p = packP(g3_wh, 256, 768, 768, 256, 768);
    _Float16* outp = packP(out_w, 256, 40, 40, 256, 48);

    // ---- activation buffers ----
    _Float16* X1   = (_Float16*)alloc((size_t)BATCH * TT * 96 * 2);
    _Float16* t1   = (_Float16*)alloc((size_t)BATCH * TT * 256 * 2);
    _Float16* y1   = (_Float16*)alloc((size_t)BATCH * (TT - 2) * 256 * 2);
    _Float16* y2   = (_Float16*)alloc((size_t)BATCH * (TT - 4) * 256 * 2);
    _Float16* cond = (_Float16*)alloc((size_t)BATCH * (TT - 4) * 256 * 2);
    _Float16* Xpre = (_Float16*)alloc((size_t)BATCH * STEPS * 352 * 2);
    float*    pre  = (float*)alloc((size_t)BATCH * STEPS * 256 * 4);

    auto blocksFor = [](long long tiles) {
        long long b = (tiles + 7) / 8;
        if (b > 4096) b = 4096;
        if (b < 1) b = 1;
        return (int)b;
    };

    // ---- conditioning network ----
    {
        int total = BATCH * TT * 96;
        build_x1<<<(total + 255) / 256, 256, 0, stream>>>(features, period,
                                                          pembed, X1);
    }
    {   // t1 = tanh(X1 @ fd1 + b)          M=6656 K=96
        long long tiles = (long long)(BATCH * TT / 16) * (256 / 16);
        gemm_wmma<1, 1, 0><<<blocksFor(tiles), 256, 0, stream>>>(
            X1, BATCH * TT, 0LL, 96, fd1p, 96, fd1_b, BATCH * TT, 256, t1, 256);
    }
    {   // y1 = tanh(conv1(t1))             M=6528 K=768 (contig windows)
        long long tiles = (long long)(BATCH * (TT - 2) / 16) * (256 / 16);
        gemm_wmma<1, 1, 0><<<blocksFor(tiles), 256, 0, stream>>>(
            t1, TT - 2, (long long)TT * 256, 256, c1p, 768, c1_b,
            BATCH * (TT - 2), 256, y1, 256);
    }
    {   // y2 = tanh(conv2(y1))             M=6400 K=768
        long long tiles = (long long)(BATCH * (TT - 4) / 16) * (256 / 16);
        gemm_wmma<1, 1, 0><<<blocksFor(tiles), 256, 0, stream>>>(
            y1, TT - 4, (long long)(TT - 2) * 256, 256, c2p, 768, c2_b,
            BATCH * (TT - 4), 256, y2, 256);
    }
    {   // cond = tanh(y2 @ fd2 + b)        M=6400 K=256
        long long tiles = (long long)(BATCH * (TT - 4) / 16) * (256 / 16);
        gemm_wmma<1, 1, 0><<<blocksFor(tiles), 256, 0, stream>>>(
            y2, BATCH * (TT - 4), 0LL, 256, fd2p, 256, fd2_b,
            BATCH * (TT - 4), 256, cond, 256);
    }

    // ---- phase embedding + Xpre assembly ----
    phase_kernel<<<BATCH, 256, 0, stream>>>(period, rshift, Xpre);
    {
        size_t total = (size_t)BATCH * STEPS * 256;
        fill_cond<<<(int)((total + 255) / 256), 256, 0, stream>>>(cond, Xpre);
    }
    {   // pre = Xpre @ Wpre + sd1_b (f32, step-major)   M=25600 K=352
        long long tiles = (long long)(BATCH * STEPS / 16) * (256 / 16);
        gemm_wmma<0, 0, 1><<<blocksFor(tiles), 256, 0, stream>>>(
            Xpre, BATCH * STEPS, 0LL, 352, prep, 352, sd1_b, BATCH * STEPS,
            256, pre, 256);
    }

    // ---- persistent recurrent loop (single WGP) ----
    recurrent_kernel<<<1, 512, 0, stream>>>(
        wpp, sd2p, wi0p, wh0p, wi1p, wh1p, wi2p, wh2p, outp, pre, sd2_b,
        g1_bi, g1_bh, g2_bi, g2_bh, g3_bi, g3_bh, out_b, out);
}
